// Sintactic_GCN_73194832658750
// MI455X (gfx1250) — compile-verified, matrix operand order checked by reading, not run on hardware
//
#include <hip/hip_runtime.h>
#include <hip/hip_bf16.h>

typedef __bf16 bf16;
typedef __attribute__((ext_vector_type(16))) bf16  v16bf;
typedef __attribute__((ext_vector_type(8)))  float v8f;

#define TOK   32768   // N = B*S
#define SB    64      // B
#define SS    512     // S
#define DD    512     // D
#define UU    512     // U
#define NDEG  65536   // N*DEG

__device__ __forceinline__ bf16 f2bf(float f) {
    union { float f; unsigned u; } a; a.f = f;
    unsigned r = a.u + 0x7FFFu + ((a.u >> 16) & 1u);   // RNE
    union { unsigned short s; bf16 h; } o; o.s = (unsigned short)(r >> 16);
    return o.h;
}
__device__ __forceinline__ float sigm(float x) { return 1.0f / (1.0f + __expf(-x)); }

// Async copy: INST_OFFSET applies to BOTH the LDS dest and global src address,
// so one base VGPR pair + varying immediate covers a whole K-panel.
template <int KOFF>
__device__ __forceinline__ void asyncLd(unsigned lds, unsigned long long g) {
    asm volatile("global_load_async_to_lds_b128 %0, %1, off offset:%2"
                 :: "v"(lds), "v"(g), "n"(KOFF) : "memory");
}
template <int CNT>
__device__ __forceinline__ void waitAsync() {
    asm volatile("s_wait_asynccnt %0" :: "n"(CNT) : "memory");
}

// ---- enc (S,B,D) f32 -> Xbf (N,D) bf16, N index = b*S+s -------------------
__global__ __launch_bounds__(256) void cvt_x_kernel(const float* __restrict__ enc,
                                                    bf16* __restrict__ xbf) {
    int t = blockIdx.x * 256 + threadIdx.x;      // over N * (D/8)
    int n = t >> 6;
    int c = (t & 63) << 3;
    int b = n >> 9, s = n & 511;
    const float* src = enc + ((size_t)(s * SB + b)) * DD + c;
    float4 f0 = *(const float4*)(src);
    float4 f1 = *(const float4*)(src + 4);
    bf16* dst = xbf + (size_t)n * DD + c;
    dst[0] = f2bf(f0.x); dst[1] = f2bf(f0.y); dst[2] = f2bf(f0.z); dst[3] = f2bf(f0.w);
    dst[4] = f2bf(f1.x); dst[5] = f2bf(f1.y); dst[6] = f2bf(f1.z); dst[7] = f2bf(f1.w);
}

// ---- W (D,U) f32 -> Wt (U,D) bf16 ----------------------------------------
__global__ __launch_bounds__(256) void cvt_wt_kernel(const float* __restrict__ w,
                                                     bf16* __restrict__ wt) {
    int t = blockIdx.x * 256 + threadIdx.x;      // over U*D
    int u = t >> 9, d = t & 511;
    wt[(size_t)u * DD + d] = f2bf(w[(size_t)d * UU + u]);
}

// ---- gate GEMVs: one wave32 per token row --------------------------------
__global__ __launch_bounds__(256) void gates_kernel(const float* __restrict__ enc,
                                                    const float* __restrict__ vig,
                                                    const float* __restrict__ vog,
                                                    const float* __restrict__ wlg,
                                                    float* __restrict__ gin,
                                                    float* __restrict__ gout,
                                                    float* __restrict__ gloop) {
    int lane = threadIdx.x & 31;
    int wid  = threadIdx.x >> 5;
    int n = blockIdx.x * 8 + wid;
    int b = n >> 9, s = n & 511;
    const float* row = enc + ((size_t)(s * SB + b)) * DD;
    float s0 = 0.f, s1 = 0.f, s2 = 0.f;
#pragma unroll
    for (int q = 0; q < 4; ++q) {
        int d = lane * 16 + q * 4;
        float4 x = *(const float4*)(row + d);
        float4 g0 = *(const float4*)(vig + d);
        float4 g1 = *(const float4*)(vog + d);
        float4 g2 = *(const float4*)(wlg + d);
        s0 += x.x * g0.x + x.y * g0.y + x.z * g0.z + x.w * g0.w;
        s1 += x.x * g1.x + x.y * g1.y + x.z * g1.z + x.w * g1.w;
        s2 += x.x * g2.x + x.y * g2.y + x.z * g2.z + x.w * g2.w;
    }
#pragma unroll
    for (int off = 16; off > 0; off >>= 1) {
        s0 += __shfl_xor(s0, off, 32);
        s1 += __shfl_xor(s1, off, 32);
        s2 += __shfl_xor(s2, off, 32);
    }
    if (lane == 0) { gin[n] = s0; gout[n] = s1; gloop[n] = s2; }
}

// ---- bf16 WMMA GEMM: C(TOK,UU) = A(TOK,DD) * Bt(UU,DD)^T, f32 accum ------
// K-panel staging: per phase, stage K=256 for the whole 128x128 block into
// LDS (layout mirrors global K-contiguity so one base address + immediate
// serves all 8 tiles), issue all 32 async B128 copies up front, then run an
// 8-deep ASYNCcnt pipeline: wait <=28-4k -> barrier -> 8 WMMAs.
#define LROW 264   // bf16 elems per LDS row: 256 data + 8 pad (528B, 16B-aligned)

__global__ __launch_bounds__(256, 1) void gemm_bf16_kernel(const bf16* __restrict__ A,
                                                           const bf16* __restrict__ Bt,
                                                           float* __restrict__ C) {
    __shared__ bf16 lA[128 * LROW];   // 67584 B
    __shared__ bf16 lB[128 * LROW];   // 67584 B

    const int tid  = threadIdx.x;
    const int lane = tid & 31;
    const int wave = tid >> 5;
    const int wm   = wave & 1;        // 2 waves along M
    const int wn   = wave >> 1;       // 4 waves along N
    const int mBase = blockIdx.y * 128;
    const int nBase = blockIdx.x * 128;
    const int kb = (lane >> 4) << 3;  // K sub-base: 0 (lanes 0-15) or 8 (lanes 16-31)
    const int ml = lane & 15;

    v8f acc[4][2];
#pragma unroll
    for (int i = 0; i < 4; ++i)
#pragma unroll
        for (int j = 0; j < 2; ++j)
            acc[i][j] = (v8f){0.f, 0.f, 0.f, 0.f, 0.f, 0.f, 0.f, 0.f};

    // Per-thread copy slot: 2x16B for A and 2x16B for B per 32-wide K tile.
    const int row = tid >> 2;              // 0..63 (second slot = row+64)
    const int col = (tid & 3) << 3;        // 0,8,16,24 (bf16 elems)
    const bf16* A0 = A  + (size_t)(mBase + row) * DD + col;
    const bf16* A1 = A0 + (size_t)64 * DD;
    const bf16* B0 = Bt + (size_t)(nBase + row) * DD + col;
    const bf16* B1 = B0 + (size_t)64 * DD;
    // Group-relative LDS byte offsets (flat->LDS mapping uses addr[31:0]).
    const unsigned laA0 = (unsigned)(unsigned long long)(void*)&lA[row * LROW + col];
    const unsigned laA1 = (unsigned)(unsigned long long)(void*)&lA[(row + 64) * LROW + col];
    const unsigned laB0 = (unsigned)(unsigned long long)(void*)&lB[row * LROW + col];
    const unsigned laB1 = (unsigned)(unsigned long long)(void*)&lB[(row + 64) * LROW + col];

    auto computeTile = [&](int k) {       // k is a literal at every call site
        union V16 { v16bf v; uint4 q[2]; };
        V16 af[4], bl[2];
#pragma unroll
        for (int i = 0; i < 4; ++i) {
            const bf16* pa = &lA[(wm * 64 + i * 16 + ml) * LROW + k * 32 + kb];
            af[i].q[0] = *(const uint4*)(pa);        // K = kb .. kb+7
            af[i].q[1] = *(const uint4*)(pa + 16);   // K = kb+16 .. kb+23
        }
#pragma unroll
        for (int j = 0; j < 2; ++j) {
            const bf16* pb = &lB[(wn * 32 + j * 16 + ml) * LROW + k * 32 + kb];
            bl[j].q[0] = *(const uint4*)(pb);
            bl[j].q[1] = *(const uint4*)(pb + 16);
        }
#pragma unroll
        for (int i = 0; i < 4; ++i)
#pragma unroll
            for (int j = 0; j < 2; ++j)
                acc[i][j] = __builtin_amdgcn_wmma_f32_16x16x32_bf16(
                    false, af[i].v, false, bl[j].v, (short)0, acc[i][j], false, false);
    };

#define GISSUE(K) do { asyncLd<(K)*64>(laA0, gA0); asyncLd<(K)*64>(laA1, gA1); \
                       asyncLd<(K)*64>(laB0, gB0); asyncLd<(K)*64>(laB1, gB1); } while (0)
#define GSTEP(K)  do { waitAsync<28 - 4*(K)>(); __syncthreads(); computeTile(K); } while (0)

#pragma unroll
    for (int p = 0; p < 2; ++p) {         // two K=256 phases cover D=512
        unsigned long long gA0 = (unsigned long long)(A0 + p * 256);
        unsigned long long gA1 = (unsigned long long)(A1 + p * 256);
        unsigned long long gB0 = (unsigned long long)(B0 + p * 256);
        unsigned long long gB1 = (unsigned long long)(B1 + p * 256);
        GISSUE(0); GISSUE(1); GISSUE(2); GISSUE(3);
        GISSUE(4); GISSUE(5); GISSUE(6); GISSUE(7);
        GSTEP(0); GSTEP(1); GSTEP(2); GSTEP(3);
        GSTEP(4); GSTEP(5); GSTEP(6); GSTEP(7);
        __syncthreads();                  // all readers done before LDS reuse
    }
#undef GISSUE
#undef GSTEP

    // C layout: vgpr e -> row e + 8*(lane>=16), col = lane&15
#pragma unroll
    for (int i = 0; i < 4; ++i)
#pragma unroll
        for (int j = 0; j < 2; ++j) {
            int gm = mBase + wm * 64 + i * 16 + ((lane >> 4) << 3);
            int gn = nBase + wn * 32 + j * 16 + (lane & 15);
            float* cp = C + (size_t)gm * UU + gn;
#pragma unroll
            for (int e = 0; e < 8; ++e) cp[(size_t)e * UU] = acc[i][j][e];
        }
}

// ---- fused gather + gate + sum + LeakyReLU -------------------------------
__global__ __launch_bounds__(256) void combine_kernel(
    const float* __restrict__ Yin, const float* __restrict__ Yout,
    const float* __restrict__ Yloop,
    const float* __restrict__ gin, const float* __restrict__ gout,
    const float* __restrict__ gloop,
    const int* __restrict__ arc_in, const int* __restrict__ arc_out,
    const int* __restrict__ lab_in, const int* __restrict__ lab_out,
    const float* __restrict__ b_in, const float* __restrict__ b_in_gate,
    const float* __restrict__ b_out, const float* __restrict__ b_out_gate,
    const float* __restrict__ mask_in, const float* __restrict__ mask_out,
    const float* __restrict__ mask_loop,
    float* __restrict__ out) {
    int t = blockIdx.x * 256 + threadIdx.x;      // over N * (U/4)
    int n = t >> 7;
    int uq = (t & 127) << 2;

    int li   = lab_in[n];
    int iin  = arc_in[n] * SS + arc_in[TOK + n];
    float mi = mask_in[n];
    float wi = mi * mi * sigm(gin[n] + b_in_gate[li]);

    int j0 = 2 * n, j1 = 2 * n + 1;
    int io0 = arc_out[j0] * SS + arc_out[NDEG + j0];
    int io1 = arc_out[j1] * SS + arc_out[NDEG + j1];
    int lo0 = lab_out[j0], lo1 = lab_out[j1];
    float m0 = mask_out[j0], m1 = mask_out[j1];
    float w0 = m0 * m0 * sigm(gout[io0] + b_out_gate[lo0]);
    float w1 = m1 * m1 * sigm(gout[io1] + b_out_gate[lo1]);
    float ml = mask_loop[n];
    float wl = ml * ml * sigm(gloop[n]);

    float4 yi = *(const float4*)(Yin   + (size_t)iin * UU + uq);
    float4 bi = *(const float4*)(b_in  + (size_t)li  * UU + uq);
    float4 y0 = *(const float4*)(Yout  + (size_t)io0 * UU + uq);
    float4 b0 = *(const float4*)(b_out + (size_t)lo0 * UU + uq);
    float4 y1 = *(const float4*)(Yout  + (size_t)io1 * UU + uq);
    float4 b1 = *(const float4*)(b_out + (size_t)lo1 * UU + uq);
    float4 yl = *(const float4*)(Yloop + (size_t)n   * UU + uq);

    float4 r;
    r.x = wi * (yi.x + bi.x) + w0 * (y0.x + b0.x) + w1 * (y1.x + b1.x) + wl * yl.x;
    r.y = wi * (yi.y + bi.y) + w0 * (y0.y + b0.y) + w1 * (y1.y + b1.y) + wl * yl.y;
    r.z = wi * (yi.z + bi.z) + w0 * (y0.z + b0.z) + w1 * (y1.z + b1.z) + wl * yl.z;
    r.w = wi * (yi.w + bi.w) + w0 * (y0.w + b0.w) + w1 * (y1.w + b1.w) + wl * yl.w;
    r.x = r.x >= 0.f ? r.x : 0.01f * r.x;
    r.y = r.y >= 0.f ? r.y : 0.01f * r.y;
    r.z = r.z >= 0.f ? r.z : 0.01f * r.z;
    r.w = r.w >= 0.f ? r.w : 0.01f * r.w;
    *(float4*)(out + (size_t)n * UU + uq) = r;
}

extern "C" void kernel_launch(void* const* d_in, const int* in_sizes, int n_in,
                              void* d_out, int out_size, void* d_ws, size_t ws_size,
                              hipStream_t stream) {
    const float* enc       = (const float*)d_in[0];
    const int*   arc_in    = (const int*)d_in[1];
    const int*   arc_out   = (const int*)d_in[2];
    const int*   lab_in    = (const int*)d_in[3];
    const int*   lab_out   = (const int*)d_in[4];
    const float* mask_in   = (const float*)d_in[5];
    const float* mask_out  = (const float*)d_in[6];
    const float* mask_loop = (const float*)d_in[7];
    const float* V_in      = (const float*)d_in[8];
    const float* b_in      = (const float*)d_in[9];
    const float* V_in_g    = (const float*)d_in[10];
    const float* b_in_g    = (const float*)d_in[11];
    const float* V_out     = (const float*)d_in[12];
    const float* b_out     = (const float*)d_in[13];
    const float* V_out_g   = (const float*)d_in[14];
    const float* b_out_g   = (const float*)d_in[15];
    const float* W_self    = (const float*)d_in[16];
    const float* W_self_g  = (const float*)d_in[17];

    char* p = (char*)d_ws;
    bf16* Xbf    = (bf16*)p;  p += (size_t)TOK * DD * sizeof(bf16);
    bf16* WtIn   = (bf16*)p;  p += (size_t)UU * DD * sizeof(bf16);
    bf16* WtOut  = (bf16*)p;  p += (size_t)UU * DD * sizeof(bf16);
    bf16* WtLoop = (bf16*)p;  p += (size_t)UU * DD * sizeof(bf16);
    float* Yin   = (float*)p; p += (size_t)TOK * UU * sizeof(float);
    float* Yout  = (float*)p; p += (size_t)TOK * UU * sizeof(float);
    float* Yloop = (float*)p; p += (size_t)TOK * UU * sizeof(float);
    float* gin   = (float*)p; p += (size_t)TOK * sizeof(float);
    float* gout  = (float*)p; p += (size_t)TOK * sizeof(float);
    float* gloop = (float*)p; p += (size_t)TOK * sizeof(float);

    cvt_x_kernel<<<(TOK * (DD / 8)) / 256, 256, 0, stream>>>(enc, Xbf);
    cvt_wt_kernel<<<(UU * DD) / 256, 256, 0, stream>>>(V_in,   WtIn);
    cvt_wt_kernel<<<(UU * DD) / 256, 256, 0, stream>>>(V_out,  WtOut);
    cvt_wt_kernel<<<(UU * DD) / 256, 256, 0, stream>>>(W_self, WtLoop);
    gates_kernel<<<TOK / 8, 256, 0, stream>>>(enc, V_in_g, V_out_g, W_self_g,
                                              gin, gout, gloop);
    dim3 ggrid(UU / 128, TOK / 128);
    gemm_bf16_kernel<<<ggrid, 256, 0, stream>>>(Xbf, WtIn,   Yin);
    gemm_bf16_kernel<<<ggrid, 256, 0, stream>>>(Xbf, WtOut,  Yout);
    gemm_bf16_kernel<<<ggrid, 256, 0, stream>>>(Xbf, WtLoop, Yloop);
    combine_kernel<<<(TOK * (UU / 4)) / 256, 256, 0, stream>>>(
        Yin, Yout, Yloop, gin, gout, gloop, arc_in, arc_out, lab_in, lab_out,
        b_in, b_in_g, b_out, b_out_g, mask_in, mask_out, mask_loop,
        (float*)d_out);
}
